// ReLUNetworkExpected_9096740733347
// MI455X (gfx1250) — matrix operand order repaired and checked by reading, not compiled, not run on hardware
//
#include <hip/hip_runtime.h>

typedef __bf16 bf16_t;
typedef bf16_t v16bf __attribute__((ext_vector_type(16)));
typedef bf16_t v8bf  __attribute__((ext_vector_type(8)));
typedef float  v8f   __attribute__((ext_vector_type(8)));

namespace {

constexpr int kB    = 2048;  // batch
constexpr int kS    = 512;   // scenarios
constexpr int kSSIn = 64;    // scenario input dim
constexpr int kSSH  = 128;   // scenario hidden dim
constexpr int kE1   = 64;    // embed dim 1
constexpr int kE2   = 128;   // embed dim 2
constexpr int kFS   = 128;   // flat features
constexpr int kRH   = 256;   // readout hidden

constexpr int kAStride  = kSSIn + 8;  // 72 bf16 -> 144B rows, bank-conflict free
constexpr int kH0Stride = kSSH + 8;   // 136 bf16 -> 272B rows, bank-conflict free

__device__ inline v16bf cat8(v8bf lo, v8bf hi) {
  return __builtin_shufflevector(lo, hi, 0, 1, 2, 3, 4, 5, 6, 7,
                                 8, 9, 10, 11, 12, 13, 14, 15);
}

__device__ inline v8f wmma_bf16(v16bf a, v16bf b, v8f c) {
  // D = A(16x32) * B(32x16) + C, f32 accumulate
  return __builtin_amdgcn_wmma_f32_16x16x32_bf16(false, a, false, b,
                                                 (short)0, c, false, false);
}

__global__ __launch_bounds__(256) void fused_scen_mlp(
    const float* __restrict__ x_fs,    // [B,128]
    const float* __restrict__ x_scen,  // [B,S,64]
    const int*   __restrict__ x_n,     // [B]
    const float* __restrict__ Ws,      // [64,128]
    const float* __restrict__ W1,      // [128,64]
    const float* __restrict__ W2,      // [64,128]
    const float* __restrict__ b2,      // [128]
    const float* __restrict__ Wr,      // [256,256]
    const float* __restrict__ br,      // [256]
    const float* __restrict__ Wo,      // [256]
    const float* __restrict__ bo,      // [1]
    float* __restrict__ out)           // [B]
{
  __shared__ __align__(16) bf16_t a_lds[32 * kAStride];    // 4.5 KB
  __shared__ __align__(16) bf16_t h0_lds[32 * kH0Stride];  // 8.5 KB
  __shared__ float agg_lds[kE1];
  __shared__ float xrow[kFS + kE2];
  __shared__ float out_acc;

  const int b    = blockIdx.x;
  const int tid  = threadIdx.x;
  const int lane = tid & 31;
  const int wave = tid >> 5;
  const int mt   = wave >> 2;        // M-tile (scenario group of 16): 0..1
  const int ntp  = (wave & 3) * 2;   // GEMM1 N-tile pair base: 0,2,4,6
  const int nt2  = wave & 3;         // GEMM2 N-tile: 0..3
  const int m    = lane & 15;        // A-fragment row within tile
  const int hi8  = (lane >> 4) * 8;  // K / M offset for upper half-wave

  // ---- Preload weight B-fragments into registers (ISA B-layout:
  //      lane L holds column N = L&15, K = 32*ks + hi8 + {0..7, 16..23}) ----
  v16bf wsf[2][2];  // [local N-tile][k-step], Ws: K=64 -> 2 k-steps
#pragma unroll
  for (int j = 0; j < 2; ++j) {
#pragma unroll
    for (int ks = 0; ks < 2; ++ks) {
      const float* p = Ws + (size_t)(ks * 32 + hi8) * kSSH + (ntp + j) * 16 + m;
      v16bf f;
#pragma unroll
      for (int e = 0; e < 16; ++e) f[e] = (bf16_t)p[(e < 8 ? e : e + 8) * kSSH];
      wsf[j][ks] = f;
    }
  }
  v16bf w1f[4];  // W1: K=128 -> 4 k-steps, single N-tile per wave
#pragma unroll
  for (int ks = 0; ks < 4; ++ks) {
    const float* p = W1 + (size_t)(ks * 32 + hi8) * kE1 + nt2 * 16 + m;
    v16bf f;
#pragma unroll
    for (int e = 0; e < 16; ++e) f[e] = (bf16_t)p[(e < 8 ? e : e + 8) * kE1];
    w1f[ks] = f;
  }

  if (tid < kE1) agg_lds[tid] = 0.0f;
  if (tid == 0) out_acc = 0.0f;

  const int n     = x_n[b];             // 1..S, block-uniform
  const int iters = (n + 31) >> 5;      // ragged early exit
  const float* const scen_base = x_scen + (size_t)b * kS * kSSIn;

  float acc = 0.0f;  // per-lane partial of agg column (nt2*16 + m)

  for (int it = 0; it < iters; ++it) {
    const int s0 = it * 32;
    // ---- Stage A tile [32 x 64] f32 -> bf16 LDS (coalesced) ----
    {
      const float* src = scen_base + (size_t)s0 * kSSIn;
      if (it + 1 < iters)  // gfx1250 global_prefetch of next tile
        __builtin_prefetch(src + 32 * kSSIn + tid * 8, 0, 0);
#pragma unroll
      for (int c = 0; c < 2; ++c) {
        const int idx = tid + c * 256;         // 512 float4 chunks
        const int row = idx >> 4;
        const int col = (idx & 15) * 4;
        const float4 v = *reinterpret_cast<const float4*>(src + row * kSSIn + col);
        bf16_t* dst = a_lds + row * kAStride + col;
        dst[0] = (bf16_t)v.x; dst[1] = (bf16_t)v.y;
        dst[2] = (bf16_t)v.z; dst[3] = (bf16_t)v.w;
      }
    }
    __syncthreads();

    // ---- GEMM1: h0 = relu(A[16x64] @ Ws[64x128]), 2 N-tiles per wave ----
    // A-fragment (ISA layout): lane holds row M=m, K = ks*32 + hi8 + {0..7, 16..23}
    v16bf a0, a1;
    {
      const bf16_t* base = a_lds + (mt * 16 + m) * kAStride + hi8;
      a0 = cat8(*(const v8bf*)(base +  0), *(const v8bf*)(base + 16));
      a1 = cat8(*(const v8bf*)(base + 32), *(const v8bf*)(base + 48));
    }
#pragma unroll
    for (int j = 0; j < 2; ++j) {
      v8f c = {};
      c = wmma_bf16(a0, wsf[j][0], c);
      c = wmma_bf16(a1, wsf[j][1], c);
      // C layout: VGPR r, lane L -> M = r + 8*(L>=16), N = L&15
      const int col = (ntp + j) * 16 + m;
#pragma unroll
      for (int r = 0; r < 8; ++r)
        h0_lds[(mt * 16 + r + hi8) * kH0Stride + col] = (bf16_t)fmaxf(c[r], 0.0f);
    }
    __syncthreads();

    // ---- GEMM2: h1 = relu(h0[16x128] @ W1[128x64]), 1 N-tile per wave ----
    v8f c2 = {};
#pragma unroll
    for (int ks = 0; ks < 4; ++ks) {
      const bf16_t* base = h0_lds + (mt * 16 + m) * kH0Stride + ks * 32 + hi8;
      v16bf af = cat8(*(const v8bf*)(base + 0), *(const v8bf*)(base + 16));
      c2 = wmma_bf16(af, w1f[ks], c2);
    }
    // ---- masked accumulate into agg partials (register-resident) ----
#pragma unroll
    for (int r = 0; r < 8; ++r) {
      const float v = fmaxf(c2[r], 0.0f);
      const int s = s0 + mt * 16 + r + hi8;
      acc += (s < n) ? v : 0.0f;
    }
    __syncthreads();  // protect a_lds / h0_lds reuse next iteration
  }

  // ---- flush agg partials (4 contributors per column: 2 halves x 2 mt) ----
  atomicAdd(&agg_lds[nt2 * 16 + m], acc);
  __syncthreads();
  if (tid < kE1) agg_lds[tid] *= (1.0f / (float)n);  // masked mean
  __syncthreads();

  // ---- emb = relu(agg @ W2 + b2); xrow = concat(x_fs[b], emb) ----
  if (tid < kE2) {
    float s = b2[tid];
#pragma unroll 8
    for (int k = 0; k < kE1; ++k) s += agg_lds[k] * W2[(size_t)k * kE2 + tid];
    xrow[kFS + tid] = fmaxf(s, 0.0f);
  } else {
    const int j = tid - kE2;
    xrow[j] = x_fs[(size_t)b * kFS + j];
  }
  __syncthreads();

  // ---- hidden = relu(xrow @ Wr + br); out = hidden @ Wo + bo ----
  float h = br[tid];
#pragma unroll 8
  for (int k = 0; k < kRH; ++k) h += xrow[k] * Wr[(size_t)k * kRH + tid];
  h = fmaxf(h, 0.0f);
  float contrib = h * Wo[tid];
#pragma unroll
  for (int off = 16; off > 0; off >>= 1)
    contrib += __shfl_down(contrib, off, 32);
  if (lane == 0) atomicAdd(&out_acc, contrib);
  __syncthreads();
  if (tid == 0) out[b] = out_acc + bo[0];
}

}  // namespace

extern "C" void kernel_launch(void* const* d_in, const int* in_sizes, int n_in,
                              void* d_out, int out_size, void* d_ws, size_t ws_size,
                              hipStream_t stream) {
  (void)in_sizes; (void)n_in; (void)out_size; (void)d_ws; (void)ws_size;
  const float* x_fs   = (const float*)d_in[0];
  const float* x_scen = (const float*)d_in[1];
  const int*   x_n    = (const int*)d_in[2];
  const float* Ws     = (const float*)d_in[3];
  const float* W1     = (const float*)d_in[4];
  const float* W2     = (const float*)d_in[5];
  const float* b2     = (const float*)d_in[6];
  const float* Wr     = (const float*)d_in[7];
  const float* br     = (const float*)d_in[8];
  const float* Wo     = (const float*)d_in[9];
  const float* bo     = (const float*)d_in[10];
  float* out = (float*)d_out;

  fused_scen_mlp<<<kB, 256, 0, stream>>>(x_fs, x_scen, x_n, Ws, W1, W2, b2,
                                         Wr, br, Wo, bo, out);
}
